// GCN2Layer_TG_996432412810
// MI455X (gfx1250) — compile-verified, matrix operand order checked by reading, not run on hardware
//
#include <hip/hip_runtime.h>
#include <hip/hip_bf16.h>

typedef __attribute__((ext_vector_type(16))) __bf16 v16bf;
typedef __attribute__((ext_vector_type(8)))  float  v8f;

__device__ __forceinline__ unsigned short f2bf(float f) {
    unsigned int u = __float_as_uint(f);
    unsigned int r = u + 0x7FFFu + ((u >> 16) & 1u);   // round-to-nearest-even
    return (unsigned short)(r >> 16);
}

// ---------------- element-wise prep kernels ----------------

__global__ void k_f32_to_bf16(const float* __restrict__ in,
                              unsigned short* __restrict__ out, int n) {
    int i = blockIdx.x * blockDim.x + threadIdx.x;
    if (i < n) out[i] = f2bf(in[i]);
}

// out[c*rows + r] = bf16(in[r*cols + c])   (store B transposed, row-major)
__global__ void k_transpose_to_bf16(const float* __restrict__ in,
                                    unsigned short* __restrict__ out,
                                    int rows, int cols) {
    int i = blockIdx.x * blockDim.x + threadIdx.x;
    if (i >= rows * cols) return;
    int r = i / cols, c = i % cols;
    out[c * rows + r] = f2bf(in[i]);
}

__global__ void k_fill(float* __restrict__ p, float v, int n) {
    int i = blockIdx.x * blockDim.x + threadIdx.x;
    if (i < n) p[i] = v;
}

__global__ void k_deg(const int* __restrict__ dst, float* __restrict__ deg, int E) {
    int i = blockIdx.x * blockDim.x + threadIdx.x;
    if (i < E) atomicAdd(&deg[dst[i]], 1.0f);
}

__global__ void k_rsqrt(float* __restrict__ d, int n) {
    int i = blockIdx.x * blockDim.x + threadIdx.x;
    if (i < n) d[i] = rsqrtf(d[i]);   // deg >= 1 always (self-loops)
}

// ---------------- WMMA GEMM: C[MxNn] = A[MxK](bf16) * Bt[NnxK]^T (+bias) ----------------
// Register-blocked: each wave computes a 16x64 strip (4 adjacent 16x16 N-tiles),
// loading the A fragment ONCE per 32-wide K-step and reusing it across 4
// v_wmma_f32_16x16x32_bf16 issues. M mult of 16, Nn mult of 64, K mult of 32.

__global__ void k_gemm_wmma(const unsigned short* __restrict__ A,
                            const unsigned short* __restrict__ Bt,
                            float* __restrict__ C,
                            const float* __restrict__ bias,
                            int M, int K, int Nn) {
    int wid = (blockIdx.x * blockDim.x + threadIdx.x) >> 5;
    int stripsN = Nn >> 6;                       // 64-column strips
    int total   = (M >> 4) * stripsN;
    if (wid >= total) return;                    // wave-uniform: EXEC all-ones inside
    int tm = wid / stripsN, sn = wid % stripsN;
    int lane = threadIdx.x & 31;
    int lo = lane & 15, hi = lane >> 4;

    const unsigned short* arow = A  + (size_t)(tm * 16 + lo) * K + hi * 8;
    const unsigned short* b0   = Bt + (size_t)(sn * 64 + lo) * K + hi * 8;

    v8f acc[4] = {{}, {}, {}, {}};
    for (int kb = 0; kb < K; kb += 32) {
        union { v16bf v; uint4 q[2]; } a;
        a.q[0] = *(const uint4*)(arow + kb);        // K = kb+8*hi    .. +8
        a.q[1] = *(const uint4*)(arow + kb + 16);   // K = kb+16+8*hi .. +8
#pragma unroll
        for (int j = 0; j < 4; ++j) {
            union { v16bf v; uint4 q[2]; } b;
            const unsigned short* br = b0 + (size_t)j * 16 * K;
            b.q[0] = *(const uint4*)(br + kb);
            b.q[1] = *(const uint4*)(br + kb + 16);
            acc[j] = __builtin_amdgcn_wmma_f32_16x16x32_bf16(
                         false, a.v, false, b.v, (short)0, acc[j], false, false);
        }
    }
#pragma unroll
    for (int j = 0; j < 4; ++j) {
        int col = sn * 64 + j * 16 + lo;
        float bv = bias ? bias[col] : 0.0f;
#pragma unroll
        for (int r = 0; r < 8; ++r) {
            int row = tm * 16 + 8 * hi + r;          // C/D layout: VGPR r -> M=r(+8*hi)
            C[(size_t)row * Nn + col] = acc[j][r] + bv;
        }
    }
}

// ---------------- graph aggregation (segment_sum of h[src]*norm into agg[dst]) ----------
// One wave per edge, 128 channels = float4 per lane. f32 atomics.

__global__ void k_aggregate(const float* __restrict__ h,
                            const int* __restrict__ src,
                            const int* __restrict__ dst,
                            const float* __restrict__ dis,
                            float* __restrict__ agg, int E) {
    int wid = (blockIdx.x * blockDim.x + threadIdx.x) >> 5;
    if (wid >= E) return;
    int lane = threadIdx.x & 31;
    int s = src[wid], d = dst[wid];
    float nrm = dis[s] * dis[d];
    float4 v = *(const float4*)(h + (size_t)s * 128 + lane * 4);
    float* o = agg + (size_t)d * 128 + lane * 4;
    atomicAdd(o + 0, v.x * nrm);
    atomicAdd(o + 1, v.y * nrm);
    atomicAdd(o + 2, v.z * nrm);
    atomicAdd(o + 3, v.w * nrm);
}

// fused: out = relu(agg + h*dis[node]^2 + bias)  -> bf16 for next GEMM
__global__ void k_selfloop_bias_relu(const float* __restrict__ agg,
                                     const float* __restrict__ h,
                                     const float* __restrict__ dis,
                                     const float* __restrict__ bias,
                                     unsigned short* __restrict__ out_bf,
                                     int N, int H) {
    int i = blockIdx.x * blockDim.x + threadIdx.x;
    if (i >= N * H) return;
    int node = i / H, ch = i % H;
    float sn = dis[node]; sn *= sn;                 // self-loop norm = 1/deg
    float v = agg[i] + h[i] * sn + bias[ch];
    out_bf[i] = f2bf(fmaxf(v, 0.0f));
}

// ---------------- log_softmax over 64 classes, one wave per row, in place --------------

__global__ void k_log_softmax64(float* __restrict__ out, int N) {
    int wid = (blockIdx.x * blockDim.x + threadIdx.x) >> 5;
    if (wid >= N) return;
    int lane = threadIdx.x & 31;
    float* row = out + (size_t)wid * 64;
    float v0 = row[lane], v1 = row[lane + 32];
    float m = fmaxf(v0, v1);
    for (int off = 16; off; off >>= 1) m = fmaxf(m, __shfl_xor(m, off, 32));
    float s = expf(v0 - m) + expf(v1 - m);
    for (int off = 16; off; off >>= 1) s += __shfl_xor(s, off, 32);
    float ls = logf(s) + m;
    row[lane]      = v0 - ls;
    row[lane + 32] = v1 - ls;
}

// ---------------- host-side orchestration ----------------

extern "C" void kernel_launch(void* const* d_in, const int* in_sizes, int n_in,
                              void* d_out, int out_size, void* d_ws, size_t ws_size,
                              hipStream_t stream) {
    const float* x  = (const float*)d_in[0];
    const float* W1 = (const float*)d_in[1];
    const float* b1 = (const float*)d_in[2];
    const float* W2 = (const float*)d_in[3];
    const float* b2 = (const float*)d_in[4];
    const float* Wl = (const float*)d_in[5];
    const float* bl = (const float*)d_in[6];
    const int* eidx = (const int*)d_in[7];

    const int H  = in_sizes[2];              // 128
    const int C  = in_sizes[6];              // 64
    const int F0 = in_sizes[1] / H;          // 512
    const int N  = in_sizes[0] / F0;         // 50000
    const int E  = in_sizes[7] / 2;          // 800000
    const int* src = eidx;
    const int* dst = eidx + E;

    // workspace carve-up (256B aligned)
    char* wsp = (char*)d_ws;
    auto carve = [&](size_t bytes) {
        char* p = wsp; wsp += (bytes + 255) & ~(size_t)255; return (void*)p;
    };
    unsigned short* xb  = (unsigned short*)carve((size_t)N * F0 * 2);
    unsigned short* w1t = (unsigned short*)carve((size_t)F0 * H * 2);
    unsigned short* w2t = (unsigned short*)carve((size_t)H * H * 2);
    unsigned short* wlt = (unsigned short*)carve((size_t)H * C * 2);
    float* dis = (float*)carve((size_t)N * 4);
    float* h   = (float*)carve((size_t)N * H * 4);
    float* agg = (float*)carve((size_t)N * H * 4);
    unsigned short* x1b = (unsigned short*)carve((size_t)N * H * 2);
    unsigned short* x2b = (unsigned short*)carve((size_t)N * H * 2);

    const int T = 256;
    auto nb = [](long long work, int t) { return (unsigned)((work + t - 1) / t); };

    // precision prep
    k_f32_to_bf16<<<nb((long long)N * F0, T), T, 0, stream>>>(x, xb, N * F0);
    k_transpose_to_bf16<<<nb(F0 * H, T), T, 0, stream>>>(W1, w1t, F0, H);
    k_transpose_to_bf16<<<nb(H * H, T), T, 0, stream>>>(W2, w2t, H, H);
    k_transpose_to_bf16<<<nb(H * C, T), T, 0, stream>>>(Wl, wlt, H, C);

    // symmetric-norm degrees (self-loops folded in via init=1)
    k_fill<<<nb(N, T), T, 0, stream>>>(dis, 1.0f, N);
    k_deg<<<nb(E, T), T, 0, stream>>>(dst, dis, E);
    k_rsqrt<<<nb(N, T), T, 0, stream>>>(dis, N);

    const unsigned gemmNH = nb((long long)(N / 16) * (H / 64), 8);  // 8 waves/block
    const unsigned gemmNC = nb((long long)(N / 16) * (C / 64), 8);

    // ---- layer 1 ----
    k_gemm_wmma<<<gemmNH, T, 0, stream>>>(xb, w1t, h, nullptr, N, F0, H);
    k_fill<<<nb((long long)N * H, T), T, 0, stream>>>(agg, 0.0f, N * H);
    k_aggregate<<<nb((long long)E * 32, T), T, 0, stream>>>(h, src, dst, dis, agg, E);
    k_selfloop_bias_relu<<<nb((long long)N * H, T), T, 0, stream>>>(agg, h, dis, b1, x1b, N, H);

    // ---- layer 2 ----
    k_gemm_wmma<<<gemmNH, T, 0, stream>>>(x1b, w2t, h, nullptr, N, H, H);
    k_fill<<<nb((long long)N * H, T), T, 0, stream>>>(agg, 0.0f, N * H);
    k_aggregate<<<nb((long long)E * 32, T), T, 0, stream>>>(h, src, dst, dis, agg, E);
    k_selfloop_bias_relu<<<nb((long long)N * H, T), T, 0, stream>>>(agg, h, dis, b2, x2b, N, H);

    // ---- classifier + log_softmax ----
    k_gemm_wmma<<<gemmNC, T, 0, stream>>>(x2b, wlt, (float*)d_out, bl, N, H, C);
    k_log_softmax64<<<nb((long long)N * 32, T), T, 0, stream>>>((float*)d_out, N);
}
